// RelativePositionalMultiHeadAttention_11244224381299
// MI455X (gfx1250) — matrix operand order repaired and checked by reading, not compiled, not run on hardware
//
#include <hip/hip_runtime.h>

// ---------------------------------------------------------------------------
// Fused windowed multi-head attention for MI455X (gfx1250, wave32, WMMA).
//
// Shapes: B=4, G=1024 windows, P=64 tokens, D=256, H=8 heads, DH=32.
// One workgroup (256 threads = 8 wave32) per window; all intermediates in LDS.
// Matmuls use v_wmma_f32_16x16x32_f16 (f16 inputs, f32 accumulate).
// Prep kernels build f16 K-major weight copies + bias in WMMA C-layout in ws.
// ---------------------------------------------------------------------------

typedef __attribute__((ext_vector_type(16))) _Float16 v16h;
typedef __attribute__((ext_vector_type(8)))  _Float16 v8h;
typedef __attribute__((ext_vector_type(8)))  float    v8f;

#define DFEAT 256
#define NHEAD 8
#define DHEAD 32
#define PTOK  64

// LDS strides (in halves); all multiples of 8 so every b128 access is aligned.
#define XSTR  264   // x / attn-out rows: 256 + 8 pad
#define QKSTR 520   // q|k rows: 512 + 8 pad
#define VTSTR 72    // v^T rows: 64 + 8 pad
#define ATSTR 72    // attn rows: 64 + 8 pad

// LDS pool offsets (bytes). x buffer is dead after QKV GEMM -> alias attn-out.
#define OFF_XAO  0                      // [64][264] f16 = 33792
#define OFF_QK   33792                  // [64][520] f16 = 66560
#define OFF_VT   100352                 // [256][72] f16 = 36864
#define OFF_AT   137216                 // [8][64][72] f16 = 73728
#define SMEM_BYTES 210944

static __device__ __forceinline__ v8f wmma32(v16h a, v16h b, v8f c) {
  return __builtin_amdgcn_wmma_f32_16x16x32_f16(false, a, false, b,
                                                (short)0, c, false, false);
}

// A-matrix lane load (16x32 f16) from a row-major [M][K] array.
// lane L: row = base_row + (L&15); chunks K = k0+g*8 and k0+16+g*8, g = L>>4.
static __device__ __forceinline__ v16h
load_A(const _Float16* p0, int row, int stride, int k0, int lg) {
  const _Float16* p = p0 + row * stride + k0 + lg * 8;
  v8h lo = *(const v8h*)p;
  v8h hi = *(const v8h*)(p + 16);
  return __builtin_shufflevector(lo, hi, 0,1,2,3,4,5,6,7,8,9,10,11,12,13,14,15);
}

// B-matrix lane load (32x16 f16) from a K-major [N][K] array ("Bt").
// lane L: column n = n_base + (L&15); chunks K = k0+g*8 and k0+16+g*8, g = L>>4.
static __device__ __forceinline__ v16h
load_B(const _Float16* bt, int n, int stride, int k0, int lg) {
  const _Float16* p = bt + n * stride + k0 + lg * 8;
  v8h lo = *(const v8h*)p;
  v8h hi = *(const v8h*)(p + 16);
  return __builtin_shufflevector(lo, hi, 0,1,2,3,4,5,6,7,8,9,10,11,12,13,14,15);
}

// ------------------------- prep kernels (run once per launch) ---------------

// Wqkv [256][768] f32 -> WqkvT [768][256] f16 (K-major for WMMA B operand)
__global__ void prep_wqkvT(const float* __restrict__ W, _Float16* __restrict__ WT) {
  int i = blockIdx.x * 256 + threadIdx.x;      // 768*256 = 196608
  int n = i >> 8, k = i & 255;
  WT[n * 256 + k] = (_Float16)W[k * 768 + n];
}

// Wm [256][256] f32 -> WmT [256][256] f16
__global__ void prep_wmT(const float* __restrict__ W, _Float16* __restrict__ WT) {
  int i = blockIdx.x * 256 + threadIdx.x;      // 65536
  int n = i >> 8, k = i & 255;
  WT[n * 256 + k] = (_Float16)W[k * 256 + n];
}

// Gather rel-pos bias directly into WMMA C-register layout:
// biasP[(((h*4+mt)*4+nt)*32+lane)*8 + r] = table[rel_index[m*64+j]][h]
// with m = mt*16 + r + 8*(lane>>4), j = nt*16 + (lane&15).
__global__ void prep_bias(const float* __restrict__ table,
                          const int* __restrict__ relidx,
                          float* __restrict__ biasP) {
  int i = blockIdx.x * 256 + threadIdx.x;      // 8*4*4*32*8 = 32768
  int r    =  i        & 7;
  int lane = (i >> 3)  & 31;
  int nt   = (i >> 8)  & 3;
  int mt   = (i >> 10) & 3;
  int h    =  i >> 12;
  int m = mt * 16 + r + 8 * (lane >> 4);
  int j = nt * 16 + (lane & 15);
  biasP[i] = table[relidx[m * 64 + j] * NHEAD + h];
}

// ------------------------------- main kernel --------------------------------

__global__ __launch_bounds__(256, 1)
void attn_fused(const float* __restrict__ x,
                const float* __restrict__ bqkv,
                const float* __restrict__ bm,
                const _Float16* __restrict__ WqkvT,
                const _Float16* __restrict__ WmT,
                const float* __restrict__ biasP,
                float* __restrict__ out) {
  __shared__ __align__(16) char smem[SMEM_BYTES];
  _Float16* xao = (_Float16*)(smem + OFF_XAO);  // x, later attn-out [64][264]
  _Float16* qk  = (_Float16*)(smem + OFF_QK);   // q (cols 0..255) | k*scale (256..511)
  _Float16* vT  = (_Float16*)(smem + OFF_VT);   // v transposed [256][72]
  _Float16* at  = (_Float16*)(smem + OFF_AT);   // softmaxed probs [8][64][72]

  const int bid  = blockIdx.x;                  // window id, 0..4095
  const int tid  = threadIdx.x;
  const int wid  = tid >> 5;
  const int lane = tid & 31;
  const int lr   = lane & 15;
  const int lg   = lane >> 4;

  // ---- Phase A: x[64][256] f32 -> f16 LDS --------------------------------
  {
    int row = tid >> 2;                         // 0..63
    int c0  = (tid & 3) * 64;
    const float4* src = (const float4*)(x + (size_t)bid * (PTOK * DFEAT)
                                          + row * DFEAT + c0);
    _Float16* dst = xao + row * XSTR + c0;
#pragma unroll
    for (int i = 0; i < 16; ++i) {
      float4 a = src[i];
      dst[i * 4 + 0] = (_Float16)a.x;
      dst[i * 4 + 1] = (_Float16)a.y;
      dst[i * 4 + 2] = (_Float16)a.z;
      dst[i * 4 + 3] = (_Float16)a.w;
    }
  }
  __syncthreads();

  // ---- Phase B: qkv = x @ Wqkv + bqkv;  q,k->qk LDS (k scaled), v->vT ----
  {
    const float scale = 0.0625f;                // DFEAT^-0.5 = 1/16
    for (int nt = 0; nt < 6; ++nt) {            // 48 n-tiles / 8 waves
      const int n0 = (wid * 6 + nt) * 16;
      v8f acc[4] = {};
#pragma unroll
      for (int k = 0; k < 8; ++k) {
        const int k0 = k * 32;
        v16h Bm = load_B(WqkvT, n0 + lr, 256, k0, lg);
#pragma unroll
        for (int m = 0; m < 4; ++m) {
          v16h Am = load_A(xao, m * 16 + lr, XSTR, k0, lg);
          acc[m] = wmma32(Am, Bm, acc[m]);
        }
      }
      const float bq = bqkv[n0 + lr];
#pragma unroll
      for (int m = 0; m < 4; ++m) {
#pragma unroll
        for (int r = 0; r < 8; ++r) {
          const int row = m * 16 + r + 8 * lg;  // output row (token)
          const float v = acc[m][r] + bq;
          if (n0 < 256) {                       // q
            qk[row * QKSTR + n0 + lr] = (_Float16)v;
          } else if (n0 < 512) {                // k (pre-scaled)
            qk[row * QKSTR + n0 + lr] = (_Float16)(v * scale);
          } else {                              // v, stored transposed [d][j]
            vT[(n0 - 512 + lr) * VTSTR + row] = (_Float16)v;
          }
        }
      }
    }
  }
  __syncthreads();

  // ---- Phase C: per-head scores + bias + softmax (wave w <-> head w) -----
  {
    const int h = wid;
    v16h Areg[4], Breg[4];
#pragma unroll
    for (int mt = 0; mt < 4; ++mt)
      Areg[mt] = load_A(qk, mt * 16 + lr, QKSTR, h * DHEAD, lg);        // q_h
#pragma unroll
    for (int nt = 0; nt < 4; ++nt)
      Breg[nt] = load_B(qk, nt * 16 + lr, QKSTR, 256 + h * DHEAD, lg);  // k_h

    v8f S[4][4];
#pragma unroll
    for (int mt = 0; mt < 4; ++mt)
#pragma unroll
      for (int nt = 0; nt < 4; ++nt) {
        v8f z = {};
        S[mt][nt] = wmma32(Areg[mt], Breg[nt], z);
      }

    // add pre-gathered bias (already in C-register layout)
#pragma unroll
    for (int mt = 0; mt < 4; ++mt)
#pragma unroll
      for (int nt = 0; nt < 4; ++nt) {
        const float* bp = biasP + ((((h * 4 + mt) * 4 + nt) * 32) + lane) * 8;
        v8f bv = *(const v8f*)bp;
        S[mt][nt] += bv;
      }

    // row softmax: reduce across 4 n-tiles and 16 lanes (width-16 xor shuffles)
#pragma unroll
    for (int mt = 0; mt < 4; ++mt) {
#pragma unroll
      for (int r = 0; r < 8; ++r) {
        float mx = S[mt][0][r];
        mx = fmaxf(mx, S[mt][1][r]);
        mx = fmaxf(mx, S[mt][2][r]);
        mx = fmaxf(mx, S[mt][3][r]);
        mx = fmaxf(mx, __shfl_xor(mx, 1, 32));
        mx = fmaxf(mx, __shfl_xor(mx, 2, 32));
        mx = fmaxf(mx, __shfl_xor(mx, 4, 32));
        mx = fmaxf(mx, __shfl_xor(mx, 8, 32));
        float p0 = __expf(S[mt][0][r] - mx);
        float p1 = __expf(S[mt][1][r] - mx);
        float p2 = __expf(S[mt][2][r] - mx);
        float p3 = __expf(S[mt][3][r] - mx);
        float sum = p0 + p1 + p2 + p3;
        sum += __shfl_xor(sum, 1, 32);
        sum += __shfl_xor(sum, 2, 32);
        sum += __shfl_xor(sum, 4, 32);
        sum += __shfl_xor(sum, 8, 32);
        const float rinv = 1.0f / sum;
        const int m = mt * 16 + r + 8 * lg;
        _Float16* ar = at + (h * PTOK + m) * ATSTR + lr;
        ar[0]  = (_Float16)(p0 * rinv);
        ar[16] = (_Float16)(p1 * rinv);
        ar[32] = (_Float16)(p2 * rinv);
        ar[48] = (_Float16)(p3 * rinv);
      }
    }
  }
  // no block barrier needed: attn rows are wave-private, vT guarded above

  // ---- Phase D: out_h = attn_h[64,64] @ v_h[64,32] -> xao (attn-out) -----
  {
    const int h = wid;
    const _Float16* ah = at + h * PTOK * ATSTR;
    v8f O[4][2] = {};
#pragma unroll
    for (int ks = 0; ks < 2; ++ks) {
      const int k0 = ks * 32;
      v16h Bv[2];
#pragma unroll
      for (int n2 = 0; n2 < 2; ++n2)
        Bv[n2] = load_B(vT, h * DHEAD + n2 * 16 + lr, VTSTR, k0, lg);
#pragma unroll
      for (int mt = 0; mt < 4; ++mt) {
        v16h Am = load_A(ah, mt * 16 + lr, ATSTR, k0, lg);
        O[mt][0] = wmma32(Am, Bv[0], O[mt][0]);
        O[mt][1] = wmma32(Am, Bv[1], O[mt][1]);
      }
    }
#pragma unroll
    for (int mt = 0; mt < 4; ++mt)
#pragma unroll
      for (int n2 = 0; n2 < 2; ++n2)
#pragma unroll
        for (int r = 0; r < 8; ++r) {
          const int row = mt * 16 + r + 8 * lg;
          xao[row * XSTR + h * DHEAD + n2 * 16 + lr] = (_Float16)O[mt][n2][r];
        }
  }
  __syncthreads();

  // ---- Phase E: result = attn_out[64,256] @ Wm + bm -> global f32 --------
  {
    float* og = out + (size_t)bid * (PTOK * DFEAT);
    for (int nt = 0; nt < 2; ++nt) {            // 16 n-tiles / 8 waves
      const int n0 = (wid * 2 + nt) * 16;
      v8f acc[4] = {};
#pragma unroll
      for (int k = 0; k < 8; ++k) {
        const int k0 = k * 32;
        v16h Bm = load_B(WmT, n0 + lr, 256, k0, lg);
#pragma unroll
        for (int m = 0; m < 4; ++m) {
          v16h Am = load_A(xao, m * 16 + lr, XSTR, k0, lg);
          acc[m] = wmma32(Am, Bm, acc[m]);
        }
      }
      const float bb = bm[n0 + lr];
#pragma unroll
      for (int m = 0; m < 4; ++m)
#pragma unroll
        for (int r = 0; r < 8; ++r) {
          const int row = m * 16 + r + 8 * lg;
          og[row * DFEAT + n0 + lr] = acc[m][r] + bb;
        }
    }
  }
}

// ------------------------------- launcher -----------------------------------

extern "C" void kernel_launch(void* const* d_in, const int* in_sizes, int n_in,
                              void* d_out, int out_size, void* d_ws, size_t ws_size,
                              hipStream_t stream) {
  const float* x          = (const float*)d_in[0];
  const float* Wqkv       = (const float*)d_in[1];
  const float* bqkv       = (const float*)d_in[2];
  const float* Wm         = (const float*)d_in[3];
  const float* bm         = (const float*)d_in[4];
  const float* bias_table = (const float*)d_in[5];
  const int*   rel_index  = (const int*)d_in[6];
  float* out = (float*)d_out;

  char* ws = (char*)d_ws;
  _Float16* WqkvT = (_Float16*)(ws);            // 768*256*2 = 393216 B
  _Float16* WmT   = (_Float16*)(ws + 393216);   // 256*256*2 = 131072 B
  float*    biasP = (float*)(ws + 524288);      // 32768*4   = 131072 B

  prep_wqkvT<<<768, 256, 0, stream>>>(Wqkv, WqkvT);
  prep_wmT  <<<256, 256, 0, stream>>>(Wm, WmT);
  prep_bias <<<128, 256, 0, stream>>>(bias_table, rel_index, biasP);

  attn_fused<<<4096, 256, 0, stream>>>(x, bqkv, bm, WqkvT, WmT, biasP, out);
}
